// Normalized_LSTMCell_3753801416770
// MI455X (gfx1250) — matrix operand order repaired and checked by reading, not compiled, not run on hardware
//
#include <hip/hip_runtime.h>

typedef __attribute__((ext_vector_type(16))) __bf16 v16bf;
typedef __attribute__((ext_vector_type(8)))  __bf16 v8bf;
typedef __attribute__((ext_vector_type(8)))  float  v8f;

#define DIM   512      // I == H
#define N4H   2048     // 4*H
#define BATCH 16384
#define BM    32       // rows per block

__device__ __forceinline__ float sigmoidf(float x) {
    return 1.0f / (1.0f + __expf(-x));
}
__device__ __forceinline__ float tanh_fast(float x) {
    x = fminf(fmaxf(x, -15.0f), 15.0f);
    float e = __expf(2.0f * x);
    return (e - 1.0f) / (e + 1.0f);
}

// ---------------------------------------------------------------------------
// Prep 1: f32 weights [512,2048] (K x N row-major) -> bf16 WMMA B-fragment
// layout. fragment(kstep, ntile): 32 lanes x 16 bf16 contiguous.
//   lane L: column n = ntile*16 + (L%16), k = kstep*32 + (L/16)*16 + j.
// packed[0] = w_hh (phase 0), packed[1] = w_ih (phase 1).
// ---------------------------------------------------------------------------
__global__ void pack_weights(const float* __restrict__ w_hh,
                             const float* __restrict__ w_ih,
                             __bf16* __restrict__ packed) {
    int tid   = blockIdx.x * blockDim.x + threadIdx.x;   // 131072 threads
    int lane  = tid & 31;
    int ntile = (tid >> 5) & 127;
    int kstep = (tid >> 12) & 15;
    int mat   = (tid >> 16) & 1;
    const float* src = mat ? w_ih : w_hh;
    __bf16* dst = packed + (size_t)mat * (DIM * N4H)
                + ((size_t)(kstep * 128 + ntile) * 512 + lane * 16);
    int n  = ntile * 16 + (lane & 15);
    int kb = kstep * 32 + (lane >> 4) * 16;
#pragma unroll
    for (int j = 0; j < 16; ++j)
        dst[j] = (__bf16)src[(size_t)(kb + j) * N4H + n];
}

// Prep 2: f32 -> bf16 activation conversion (x, h0) so the main kernel can
// async-copy A tiles straight into LDS with no conversion pass.
__global__ void cvt_to_bf16(const float* __restrict__ src,
                            __bf16* __restrict__ dst) {
    size_t i = ((size_t)blockIdx.x * blockDim.x + threadIdx.x) * 4;
    float4 v = *(const float4*)(src + i);
    dst[i + 0] = (__bf16)v.x;
    dst[i + 1] = (__bf16)v.y;
    dst[i + 2] = (__bf16)v.z;
    dst[i + 3] = (__bf16)v.w;
}

// ---------------------------------------------------------------------------
// Fused: wh = h0@w_hh + bias_h ; wi = x@w_ih + bias_x (bf16 WMMA, f32 acc)
//        s  = LN(wh)*g_hh+b_hh + LN(wi)*g_ih+b_ih   (LN over 2048)
//        f,i,o,g = split(s); c1 = sig(f)*c0 + sig(i)*tanh(g)
//        h1 = sig(o)*tanh(LN(c1)*g_c + b_c)          (LN over 512)
// ---------------------------------------------------------------------------
__global__ __launch_bounds__(256)
void lstm_ln_fused(const __bf16* __restrict__ h0bf,   // [16384,512] bf16
                   const __bf16* __restrict__ xbf,    // [16384,512] bf16
                   const float* __restrict__ c0,
                   const float* __restrict__ bias_x,
                   const float* __restrict__ bias_h,
                   const float* __restrict__ g_ih, const float* __restrict__ b_ih,
                   const float* __restrict__ g_hh, const float* __restrict__ b_hh,
                   const float* __restrict__ g_c,  const float* __restrict__ b_c,
                   const __bf16* __restrict__ packB,  // [2][512*2048]
                   float* __restrict__ out_h1,
                   float* __restrict__ out_c1) {
    __shared__ __attribute__((aligned(16))) __bf16 sA[BM][DIM];      // 32 KB (reused per phase)
    __shared__ __attribute__((aligned(16))) __bf16 raw_wh[BM][N4H];  // 128 KB
    __shared__ __attribute__((aligned(16))) __bf16 raw_wi[BM][N4H];  // 128 KB
    __shared__ float sSum[2][BM], sSumSq[2][BM];
    __shared__ float sMean[2][BM], sRstd[2][BM];

    const int tid      = threadIdx.x;
    const int lane     = tid & 31;
    const int wave     = tid >> 5;
    const int rowsBase = blockIdx.x * BM;

    if (tid < 64) { (&sSum[0][0])[tid] = 0.0f; (&sSumSq[0][0])[tid] = 0.0f; }

    const int m16  = lane & 15;    // output column within 16-wide tile
    const int half = lane >> 4;
    const unsigned ldsA = (unsigned)(uintptr_t)&sA[0][0];

    for (int phase = 0; phase < 2; ++phase) {
        // ---- Stage A tile (32x512 bf16 = 32 KB) via async copy to LDS ----
        const __bf16* Asrc = (phase ? xbf : h0bf) + (size_t)rowsBase * DIM;
#pragma unroll
        for (int it = 0; it < 8; ++it) {
            unsigned byteOff = (unsigned)(tid + it * 256) * 16u;  // 16 B / thread
            unsigned long long ga = (unsigned long long)(uintptr_t)
                                    ((const char*)Asrc + byteOff);
            unsigned la = ldsA + byteOff;
            asm volatile("global_load_async_to_lds_b128 %0, %1, off"
                         :: "v"(la), "v"(ga) : "memory");
        }
        asm volatile("s_wait_asynccnt 0x0" ::: "memory");
        __syncthreads();

        const float*  bias = phase ? bias_x : bias_h;
        const __bf16* pBm  = packB + (size_t)phase * (DIM * N4H);
        __bf16*       rawp = phase ? &raw_wi[0][0] : &raw_wh[0][0];

        // 128 n-tile jobs over 8 waves; each job covers both 16-row M-tiles
        // so every B fragment feeds two WMMAs.
        for (int t = 0; t < 16; ++t) {
            int ntile = wave + 8 * t;
            const __bf16* pB = pBm + ((size_t)ntile * 512 + lane * 16);
            const __bf16* a0base = &sA[0][0]  + m16 * DIM + half * 16;
            const __bf16* a1base = &sA[16][0] + m16 * DIM + half * 16;
            v8f acc0 = {}, acc1 = {};
#pragma unroll
            for (int ks = 0; ks < 16; ++ks) {
                v16bf b = *(const v16bf*)(pB + (size_t)ks * 128 * 512);
                const __bf16* ap0 = a0base + ks * 32;
                v8bf lo0 = *(const v8bf*)(ap0);
                v8bf hi0 = *(const v8bf*)(ap0 + 8);
                v16bf a0 = __builtin_shufflevector(lo0, hi0,
                             0,1,2,3,4,5,6,7,8,9,10,11,12,13,14,15);
                const __bf16* ap1 = a1base + ks * 32;
                v8bf lo1 = *(const v8bf*)(ap1);
                v8bf hi1 = *(const v8bf*)(ap1 + 8);
                v16bf a1 = __builtin_shufflevector(lo1, hi1,
                             0,1,2,3,4,5,6,7,8,9,10,11,12,13,14,15);
                acc0 = __builtin_amdgcn_wmma_f32_16x16x32_bf16(
                           false, a0, false, b, (short)0, acc0, false, false);
                acc1 = __builtin_amdgcn_wmma_f32_16x16x32_bf16(
                           false, a1, false, b, (short)0, acc1, false, false);
            }
            // Epilogue: +bias, raw -> LDS, row sum/sumsq for LN stats.
            int ncol = ntile * 16 + m16;
            float bv = bias[ncol];
#pragma unroll
            for (int mt = 0; mt < 2; ++mt) {
#pragma unroll
                for (int r = 0; r < 8; ++r) {
                    float v  = (mt ? acc1[r] : acc0[r]) + bv;
                    int rowM = mt * 16 + half * 8 + r;
                    rawp[rowM * N4H + ncol] = (__bf16)v;
                    float sv = v, qv = v * v;
#pragma unroll
                    for (int off = 8; off >= 1; off >>= 1) {  // 16-lane groups
                        sv += __shfl_xor(sv, off, 32);
                        qv += __shfl_xor(qv, off, 32);
                    }
                    if (m16 == 0) {
                        atomicAdd(&sSum[phase][rowM], sv);
                        atomicAdd(&sSumSq[phase][rowM], qv);
                    }
                }
            }
        }
        __syncthreads();   // all waves done with sA before next phase reloads
    }

    if (tid < 64) {
        int mat = tid >> 5, mm = tid & 31;
        float mean = sSum[mat][mm] * (1.0f / (float)N4H);
        float var  = sSumSq[mat][mm] * (1.0f / (float)N4H) - mean * mean;
        sMean[mat][mm] = mean;
        sRstd[mat][mm] = rsqrtf(var + 1e-5f);
    }
    __syncthreads();

    // Elementwise tail: 4 rows per wave; lane covers cols n = lane + 32j.
    for (int rr = 0; rr < 4; ++rr) {
        int m       = wave * 4 + rr;
        size_t grow = (size_t)(rowsBase + m);
        float mu0 = sMean[0][m], r0 = sRstd[0][m];
        float mu1 = sMean[1][m], r1 = sRstd[1][m];

        auto s_at = [&](int idx) -> float {
            return ((float)raw_wh[m][idx] - mu0) * r0 * g_hh[idx] + b_hh[idx]
                 + ((float)raw_wi[m][idx] - mu1) * r1 * g_ih[idx] + b_ih[idx];
        };

        float c1v[16], ov[16];
        float csum = 0.0f, cq = 0.0f;
#pragma unroll
        for (int j = 0; j < 16; ++j) {
            int n = lane + 32 * j;
            float fs = s_at(n);
            float is = s_at(n + 512);
            float os = s_at(n + 1024);
            float gs = s_at(n + 1536);
            float c0v = __builtin_nontemporal_load(&c0[grow * DIM + n]);
            float c1  = sigmoidf(fs) * c0v + sigmoidf(is) * tanh_fast(gs);
            ov[j]  = sigmoidf(os);
            c1v[j] = c1;
            csum += c1;
            cq   += c1 * c1;
        }
#pragma unroll
        for (int off = 16; off >= 1; off >>= 1) {       // full-wave reduction
            csum += __shfl_xor(csum, off, 32);
            cq   += __shfl_xor(cq,   off, 32);
        }
        float mc = csum * (1.0f / (float)DIM);
        float vc = cq   * (1.0f / (float)DIM) - mc * mc;
        float rc = rsqrtf(vc + 1e-5f);
#pragma unroll
        for (int j = 0; j < 16; ++j) {
            int n = lane + 32 * j;
            float ln = (c1v[j] - mc) * rc * g_c[n] + b_c[n];
            __builtin_nontemporal_store(ov[j] * tanh_fast(ln),
                                        &out_h1[grow * DIM + n]);
            __builtin_nontemporal_store(c1v[j], &out_c1[grow * DIM + n]);
        }
    }
}

extern "C" void kernel_launch(void* const* d_in, const int* in_sizes, int n_in,
                              void* d_out, int out_size, void* d_ws, size_t ws_size,
                              hipStream_t stream) {
    const float* x      = (const float*)d_in[0];
    const float* h0     = (const float*)d_in[1];
    const float* c0     = (const float*)d_in[2];
    const float* w_ih   = (const float*)d_in[3];
    const float* w_hh   = (const float*)d_in[4];
    const float* bias_x = (const float*)d_in[5];
    const float* bias_h = (const float*)d_in[6];
    const float* g_ih   = (const float*)d_in[7];
    const float* b_ih   = (const float*)d_in[8];
    const float* g_hh   = (const float*)d_in[9];
    const float* b_hh   = (const float*)d_in[10];
    const float* g_c    = (const float*)d_in[11];
    const float* b_c    = (const float*)d_in[12];

    float*  out    = (float*)d_out;                 // [h1 | c1] concatenated
    // Workspace layout (bf16 elements):
    //   packB : 2 * 512*2048              =  4 MB
    //   h0bf  : 16384*512                 = 16 MB
    //   xbf   : 16384*512                 = 16 MB
    __bf16* packed = (__bf16*)d_ws;
    __bf16* h0bf   = packed + (size_t)2 * DIM * N4H;
    __bf16* xbf    = h0bf + (size_t)BATCH * DIM;

    pack_weights<<<512, 256, 0, stream>>>(w_hh, w_ih, packed);
    cvt_to_bf16 <<<(BATCH * DIM / 4) / 256, 256, 0, stream>>>(h0, h0bf);
    cvt_to_bf16 <<<(BATCH * DIM / 4) / 256, 256, 0, stream>>>(x,  xbf);

    lstm_ln_fused<<<BATCH / BM, 256, 0, stream>>>(
        h0bf, xbf, c0, bias_x, bias_h,
        g_ih, b_ih, g_hh, b_hh, g_c, b_c,
        packed, out, out + (size_t)BATCH * DIM);
}
